// get_model_61065845014592
// MI455X (gfx1250) — matrix operand order repaired and checked by reading, not compiled, not run on hardware
//
#include <hip/hip_runtime.h>

#define BB 16
#define NPTS 2048
#define GG 256
#define KK 32
#define DMODEL 384
#define NLAYER 12
#define DINNER 768
#define DSTATE 16
#define DTRANK 24
#define DCONV 4
#define LTOK 768   // N_ORDERS * G
#define EPSV 1e-5f

typedef _Float16 v16h __attribute__((ext_vector_type(16)));
typedef _Float16 v8h  __attribute__((ext_vector_type(8)));
typedef float    v8f  __attribute__((ext_vector_type(8)));

// ---------------- WMMA fragment helpers (CDNA5 16x16x32 f16, wave32) --------
// A (16x32, row-major src): lane m<16 holds row m, K = k0..k0+7 (elems 0..7)
// and K = k0+16..k0+23 (elems 8..15); lane m+16 holds the K=+8 variants.
__device__ __forceinline__ v16h ldfragA(const _Float16* A, int lda, int m0, int k0, int lane) {
  int r  = m0 + (lane & 15);
  int kk = k0 + ((lane >> 4) << 3);
  const _Float16* p = A + (size_t)r * lda + kk;
  v8h lo = *(const v8h*)(p);
  v8h hi = *(const v8h*)(p + 16);
  v16h f;
#pragma unroll
  for (int i = 0; i < 8; ++i) { f[i] = lo[i]; f[i + 8] = hi[i]; }
  return f;
}
// B (32x16) loaded from Bt = B^T (N x K row-major, i.e. weight[out][in]):
// lane n holds col n&15; lanes<16 hold K = k0..k0+7, k0+16..k0+23; lanes>=16 hold the K=+8 variants.
__device__ __forceinline__ v16h ldfragB(const _Float16* Bt, int ldb, int n0, int k0, int lane) {
  int c  = n0 + (lane & 15);
  int kk = k0 + ((lane >> 4) << 3);
  const _Float16* p = Bt + (size_t)c * ldb + kk;
  v8h lo = *(const v8h*)(p);
  v8h hi = *(const v8h*)(p + 16);
  v16h f;
#pragma unroll
  for (int i = 0; i < 8; ++i) { f[i] = lo[i]; f[i + 8] = hi[i]; }
  return f;
}
__device__ __forceinline__ v8f wmma_f16(v16h a, v16h b, v8f c) {
  return __builtin_amdgcn_wmma_f32_16x16x32_f16(false, a, false, b, (short)0, c, false, false);
}

// ---------------- small utility kernels -------------------------------------
__global__ void cvt_f16_pad(const float* __restrict__ src, _Float16* __restrict__ dst,
                            int nl, int rows, int rowsPad, int cols) {
  size_t n = (size_t)nl * rowsPad * cols;
  for (size_t i = (size_t)blockIdx.x * 256 + threadIdx.x; i < n; i += (size_t)gridDim.x * 256) {
    int c = (int)(i % cols);
    size_t t = i / cols;
    int r = (int)(t % rowsPad);
    int l = (int)(t / rowsPad);
    float v = (r < rows) ? src[((size_t)l * rows + r) * cols + c] : 0.f;
    dst[i] = (_Float16)v;
  }
}

__global__ void center_gather(const float* __restrict__ pts, const int* __restrict__ fps,
                              float* __restrict__ center) {
  int i = blockIdx.x * 256 + threadIdx.x;
  if (i >= BB * GG) return;
  int b = i / GG;
  int idx = fps[i];
  const float* p = pts + ((size_t)b * NPTS + idx) * 3;
  center[i * 3 + 0] = p[0];
  center[i * 3 + 1] = p[1];
  center[i * 3 + 2] = p[2];
}

// ---------------- KNN: one block per (b,g), iterative argmin ----------------
__global__ void __launch_bounds__(256) knn_kernel(const float* __restrict__ pts,
                                                  const float* __restrict__ center,
                                                  int* __restrict__ knn) {
  __shared__ float dist[NPTS];
  __shared__ float rmin[256];
  __shared__ int   ridx[256];
  int bg = blockIdx.x, b = bg / GG, tid = threadIdx.x;
  float c0 = center[bg * 3], c1 = center[bg * 3 + 1], c2 = center[bg * 3 + 2];
  for (int n = tid; n < NPTS; n += 256) {
    const float* p = pts + ((size_t)b * NPTS + n) * 3;
    float dx = p[0] - c0, dy = p[1] - c1, dz = p[2] - c2;
    dist[n] = dx * dx + dy * dy + dz * dz;
  }
  __syncthreads();
  for (int j = 0; j < KK; ++j) {
    float best = 3.4e38f; int bi = NPTS;
    for (int n = tid; n < NPTS; n += 256) {
      float v = dist[n];
      if (v < best) { best = v; bi = n; }
    }
    rmin[tid] = best; ridx[tid] = bi;
    __syncthreads();
    for (int s = 128; s > 0; s >>= 1) {
      if (tid < s) {
        float v = rmin[tid + s]; int ix = ridx[tid + s];
        if (v < rmin[tid] || (v == rmin[tid] && ix < ridx[tid])) { rmin[tid] = v; ridx[tid] = ix; }
      }
      __syncthreads();
    }
    if (tid == 0) { knn[bg * KK + j] = ridx[0]; dist[ridx[0]] = 3.4e38f; }
    __syncthreads();
  }
}

// ---------------- fused PointNet encoder: one block per (b,g) ---------------
__global__ void __launch_bounds__(256) encoder_kernel(
    const float* __restrict__ pts, const int* __restrict__ knn, const float* __restrict__ center,
    const float* __restrict__ w1, const float* __restrict__ b1,
    const float* __restrict__ g1, const float* __restrict__ bb1,
    const float* __restrict__ m1, const float* __restrict__ v1,
    const _Float16* __restrict__ w2h, const float* __restrict__ b2,
    const _Float16* __restrict__ w3h, const float* __restrict__ b3,
    const float* __restrict__ g2, const float* __restrict__ bb2,
    const float* __restrict__ m2, const float* __restrict__ v2,
    const _Float16* __restrict__ w4h, const float* __restrict__ b4,
    float* __restrict__ tokens) {
  __shared__ _Float16 sA[32 * 512];   // f1 (32x128), then concat (32x512)
  __shared__ _Float16 sC[32 * 512];   // f3 output f16
  __shared__ float    sF[32 * 512];   // f2 (32x256) f32, then f4 (32x384) f32
  __shared__ float    sNb[32 * 3];
  __shared__ float    sFg[256];

  int tid = threadIdx.x, lane = tid & 31, wave = tid >> 5;
  int bg = blockIdx.x, b = bg / GG;

  if (tid < 32 * 3) {
    int k = tid / 3, c = tid % 3;
    int nidx = knn[bg * KK + k];
    sNb[k * 3 + c] = pts[((size_t)b * NPTS + nidx) * 3 + c] - center[bg * 3 + c];
  }
  __syncthreads();

  // f1: 32x128 = relu(bn1(nb @ w1^T + b1))
  {
    int k = tid >> 3, o0 = (tid & 7) << 4;
    float x0 = sNb[k * 3], x1 = sNb[k * 3 + 1], x2 = sNb[k * 3 + 2];
#pragma unroll
    for (int j = 0; j < 16; ++j) {
      int o = o0 + j;
      float s = b1[o] + x0 * w1[o * 3] + x1 * w1[o * 3 + 1] + x2 * w1[o * 3 + 2];
      s = (s - m1[o]) * rsqrtf(v1[o] + EPSV) * g1[o] + bb1[o];
      sA[k * 128 + o] = (_Float16)fmaxf(s, 0.f);
    }
  }
  __syncthreads();

  // GEMM1: (32x128)x(128x256) + b2 -> sF f32 (ld 256)
  for (int t = wave; t < 32; t += 8) {
    int m0 = (t & 1) << 4, n0 = (t >> 1) << 4;
    v8f acc = {};
    for (int k0 = 0; k0 < 128; k0 += 32) {
      v16h a = ldfragA(sA, 128, m0, k0, lane);
      v16h bm = ldfragB(w2h, 128, n0, k0, lane);
      acc = wmma_f16(a, bm, acc);
    }
    int col = n0 + (lane & 15), r0 = m0 + ((lane >> 4) << 3);
    float bias = b2[col];
#pragma unroll
    for (int j = 0; j < 8; ++j) sF[(r0 + j) * 256 + col] = acc[j] + bias;
  }
  __syncthreads();

  // fg = max over K ; concat [fg | f2] as f16 in sA (32x512)
  {
    float mx = -3.4e38f;
    for (int k = 0; k < 32; ++k) mx = fmaxf(mx, sF[k * 256 + tid]);
    sFg[tid] = mx;
  }
  __syncthreads();
  for (int i = tid; i < 32 * 512; i += 256) {
    int k = i >> 9, o = i & 511;
    float v = (o < 256) ? sFg[o] : sF[k * 256 + (o - 256)];
    sA[i] = (_Float16)v;
  }
  __syncthreads();

  // GEMM2: (32x512)x(512x512) + b3, bn2, relu -> sC f16 (ld 512)
  for (int t = wave; t < 64; t += 8) {
    int m0 = (t & 1) << 4, n0 = (t >> 1) << 4;
    v8f acc = {};
    for (int k0 = 0; k0 < 512; k0 += 32) {
      v16h a = ldfragA(sA, 512, m0, k0, lane);
      v16h bm = ldfragB(w3h, 512, n0, k0, lane);
      acc = wmma_f16(a, bm, acc);
    }
    int col = n0 + (lane & 15), r0 = m0 + ((lane >> 4) << 3);
    float bias = b3[col];
    float sc = rsqrtf(v2[col] + EPSV) * g2[col];
    float mm = m2[col], bo = bb2[col];
#pragma unroll
    for (int j = 0; j < 8; ++j) {
      float s = (acc[j] + bias - mm) * sc + bo;
      sC[(r0 + j) * 512 + col] = (_Float16)fmaxf(s, 0.f);
    }
  }
  __syncthreads();

  // GEMM3: (32x512)x(512x384) + b4 -> sF f32 (ld 384)
  for (int t = wave; t < 48; t += 8) {
    int m0 = (t & 1) << 4, n0 = (t >> 1) << 4;
    v8f acc = {};
    for (int k0 = 0; k0 < 512; k0 += 32) {
      v16h a = ldfragA(sC, 512, m0, k0, lane);
      v16h bm = ldfragB(w4h, 512, n0, k0, lane);
      acc = wmma_f16(a, bm, acc);
    }
    int col = n0 + (lane & 15), r0 = m0 + ((lane >> 4) << 3);
    float bias = b4[col];
#pragma unroll
    for (int j = 0; j < 8; ++j) sF[(r0 + j) * 384 + col] = acc[j] + bias;
  }
  __syncthreads();

  for (int o = tid; o < DMODEL; o += 256) {
    float mx = -3.4e38f;
    for (int k = 0; k < 32; ++k) mx = fmaxf(mx, sF[k * 384 + o]);
    tokens[(size_t)bg * DMODEL + o] = mx;
  }
}

// ---------------- positional MLP (gelu-tanh) -------------------------------
__global__ void __launch_bounds__(128) pos_mlp_kernel(
    const float* __restrict__ center, const float* __restrict__ w1, const float* __restrict__ b1,
    const float* __restrict__ w2, const float* __restrict__ b2, float* __restrict__ pos) {
  __shared__ float ph[128];
  int bg = blockIdx.x, tid = threadIdx.x;
  float c0 = center[bg * 3], c1 = center[bg * 3 + 1], c2 = center[bg * 3 + 2];
  float s = b1[tid] + c0 * w1[tid * 3] + c1 * w1[tid * 3 + 1] + c2 * w1[tid * 3 + 2];
  float t = tanhf(0.7978845608028654f * (s + 0.044715f * s * s * s));
  ph[tid] = 0.5f * s * (1.f + t);
  __syncthreads();
  for (int o = tid; o < DMODEL; o += 128) {
    float acc = b2[o];
    const float* w = w2 + o * 128;
    for (int h = 0; h < 128; ++h) acc += ph[h] * w[h];
    pos[(size_t)bg * DMODEL + o] = acc;
  }
}

__global__ void reorder_init(const float* __restrict__ tokens, const float* __restrict__ pos,
                             const int* __restrict__ sidx, float* __restrict__ hidden,
                             float* __restrict__ resid) {
  size_t n = (size_t)BB * LTOK * DMODEL;
  for (size_t i = (size_t)blockIdx.x * 256 + threadIdx.x; i < n; i += (size_t)gridDim.x * 256) {
    int d = (int)(i % DMODEL);
    size_t t = i / DMODEL;
    int l = (int)(t % LTOK);
    int b = (int)(t / LTOK);
    int id = sidx[b * LTOK + l];
    size_t src = ((size_t)b * GG + id) * DMODEL + d;
    hidden[i] = tokens[src] + pos[src];
    resid[i] = 0.f;
  }
}

// ---------------- residual + LayerNorm (writes f16 for WMMA) ----------------
__global__ void __launch_bounds__(128) ln_residual(const float* __restrict__ hidden,
                                                   float* __restrict__ resid,
                                                   const float* __restrict__ w,
                                                   const float* __restrict__ bsh,
                                                   _Float16* __restrict__ ln16) {
  __shared__ float red[128];
  int tid = threadIdx.x;
  size_t base = (size_t)blockIdx.x * DMODEL;
  float x0 = hidden[base + tid]       + resid[base + tid];
  float x1 = hidden[base + tid + 128] + resid[base + tid + 128];
  float x2 = hidden[base + tid + 256] + resid[base + tid + 256];
  resid[base + tid] = x0; resid[base + tid + 128] = x1; resid[base + tid + 256] = x2;
  red[tid] = x0 + x1 + x2;
  __syncthreads();
  for (int s = 64; s > 0; s >>= 1) { if (tid < s) red[tid] += red[tid + s]; __syncthreads(); }
  float mean = red[0] * (1.f / DMODEL);
  __syncthreads();
  float d0 = x0 - mean, d1 = x1 - mean, d2 = x2 - mean;
  red[tid] = d0 * d0 + d1 * d1 + d2 * d2;
  __syncthreads();
  for (int s = 64; s > 0; s >>= 1) { if (tid < s) red[tid] += red[tid + s]; __syncthreads(); }
  float inv = rsqrtf(red[0] * (1.f / DMODEL) + EPSV);
  ln16[base + tid]       = (_Float16)(d0 * inv * w[tid]       + bsh[tid]);
  ln16[base + tid + 128] = (_Float16)(d1 * inv * w[tid + 128] + bsh[tid + 128]);
  ln16[base + tid + 256] = (_Float16)(d2 * inv * w[tid + 256] + bsh[tid + 256]);
}

__global__ void __launch_bounds__(128) final_ln(const float* __restrict__ hidden,
                                                const float* __restrict__ resid,
                                                const float* __restrict__ w,
                                                const float* __restrict__ bsh,
                                                float* __restrict__ out) {
  __shared__ float red[128];
  int tid = threadIdx.x;
  size_t base = (size_t)blockIdx.x * DMODEL;
  float x0 = hidden[base + tid]       + resid[base + tid];
  float x1 = hidden[base + tid + 128] + resid[base + tid + 128];
  float x2 = hidden[base + tid + 256] + resid[base + tid + 256];
  red[tid] = x0 + x1 + x2;
  __syncthreads();
  for (int s = 64; s > 0; s >>= 1) { if (tid < s) red[tid] += red[tid + s]; __syncthreads(); }
  float mean = red[0] * (1.f / DMODEL);
  __syncthreads();
  float d0 = x0 - mean, d1 = x1 - mean, d2 = x2 - mean;
  red[tid] = d0 * d0 + d1 * d1 + d2 * d2;
  __syncthreads();
  for (int s = 64; s > 0; s >>= 1) { if (tid < s) red[tid] += red[tid + s]; __syncthreads(); }
  float inv = rsqrtf(red[0] * (1.f / DMODEL) + EPSV);
  out[base + tid]       = d0 * inv * w[tid]       + bsh[tid];
  out[base + tid + 128] = d1 * inv * w[tid + 128] + bsh[tid + 128];
  out[base + tid + 256] = d2 * inv * w[tid + 256] + bsh[tid + 256];
}

// ------- register-blocked WMMA GEMM: each wave owns a 32x64 macro-tile ------
// C = A(MxK,f16,row-major) * Bt(NxK,f16,row-major)^T (+bias).
// 2 A-frags x 4 B-frags -> 8 accumulators: 6KB of loads per 8 WMMAs
// (~22 flop/byte from L2 instead of ~8 for an unblocked 16x16 tile).
__global__ void __launch_bounds__(256) gemm_f16(const _Float16* __restrict__ A, int lda,
                                                const _Float16* __restrict__ Bt, int ldb,
                                                float* __restrict__ C, int ldc,
                                                const float* __restrict__ bias,
                                                int Kd, int Nt64, int tiles) {
  int lane = threadIdx.x & 31;
  int tile = blockIdx.x * 8 + (threadIdx.x >> 5);
  if (tile >= tiles) return;
  int mt = tile / Nt64, nt = tile % Nt64;
  int m0 = mt << 5, n0 = nt << 6;
  v8f acc[2][4] = {};
  for (int k0 = 0; k0 < Kd; k0 += 32) {
    if (k0 + 64 < Kd)  // pull next B slice toward the WGP (global_prefetch_b8)
      __builtin_prefetch(Bt + (size_t)(n0 + (lane & 15)) * ldb + k0 + 64, 0, 0);
    v16h a0 = ldfragA(A, lda, m0, k0, lane);
    v16h a1 = ldfragA(A, lda, m0 + 16, k0, lane);
#pragma unroll
    for (int j = 0; j < 4; ++j) {
      v16h bm = ldfragB(Bt, ldb, n0 + 16 * j, k0, lane);
      acc[0][j] = wmma_f16(a0, bm, acc[0][j]);
      acc[1][j] = wmma_f16(a1, bm, acc[1][j]);
    }
  }
  int colb = lane & 15, r0 = (lane >> 4) << 3;
#pragma unroll
  for (int j = 0; j < 4; ++j) {
    int col = n0 + 16 * j + colb;
    float bb = bias ? bias[col] : 0.f;
#pragma unroll
    for (int i = 0; i < 2; ++i) {
      int rb = m0 + 16 * i + r0;
#pragma unroll
      for (int q = 0; q < 8; ++q) C[(size_t)(rb + q) * ldc + col] = acc[i][j][q] + bb;
    }
  }
}

// ---------------- Mamba elementwise stages ----------------------------------
__global__ void conv_silu(const float* __restrict__ xz, const float* __restrict__ cw,
                          const float* __restrict__ cb, float* __restrict__ xc,
                          _Float16* __restrict__ xc16) {
  size_t n = (size_t)BB * LTOK * DINNER;
  for (size_t i = (size_t)blockIdx.x * 256 + threadIdx.x; i < n; i += (size_t)gridDim.x * 256) {
    int d = (int)(i % DINNER);
    size_t t = i / DINNER;
    int l = (int)(t % LTOK);
    size_t b = t / LTOK;
    float s = cb[d];
#pragma unroll
    for (int k = 0; k < DCONV; ++k) {
      int ls = l - (DCONV - 1) + k;
      if (ls >= 0) s += cw[d * DCONV + k] * xz[((size_t)b * LTOK + ls) * (2 * DINNER) + d];
    }
    s = s / (1.f + expf(-s));
    xc[i] = s;
    xc16[i] = (_Float16)s;
  }
}

__global__ void dt_softplus(const float* __restrict__ dbl, const float* __restrict__ dtW,
                            const float* __restrict__ dtB, float* __restrict__ dt) {
  size_t n = (size_t)BB * LTOK * DINNER;
  for (size_t i = (size_t)blockIdx.x * 256 + threadIdx.x; i < n; i += (size_t)gridDim.x * 256) {
    int d = (int)(i % DINNER);
    size_t row = i / DINNER;
    const float* dr = dbl + row * 64;
    const float* w = dtW + (size_t)d * DTRANK;
    float s = dtB[d];
#pragma unroll
    for (int r = 0; r < DTRANK; ++r) s += dr[r] * w[r];
    dt[i] = (s > 20.f) ? s : log1pf(expf(s));
  }
}

// selective scan: 48 blocks (B x 3), each thread owns one channel's h[16]
__global__ void __launch_bounds__(256) scan_kernel(const float* __restrict__ dbl,
                                                   const float* __restrict__ dt,
                                                   const float* __restrict__ xc,
                                                   const float* __restrict__ A_log,
                                                   const float* __restrict__ Dp,
                                                   float* __restrict__ y) {
  __shared__ float sB[DSTATE], sC[DSTATE];
  int tid = threadIdx.x;
  int b = blockIdx.x / 3;
  int d = (blockIdx.x % 3) * 256 + tid;
  float A[DSTATE], h[DSTATE];
#pragma unroll
  for (int s = 0; s < DSTATE; ++s) { A[s] = -expf(A_log[(size_t)d * DSTATE + s]); h[s] = 0.f; }
  float Dd = Dp[d];
  for (int l = 0; l < LTOK; ++l) {
    size_t row = (size_t)b * LTOK + l;
    if (tid < DSTATE) sB[tid] = dbl[row * 64 + DTRANK + tid];
    else if (tid < 2 * DSTATE) sC[tid - DSTATE] = dbl[row * 64 + DTRANK + DSTATE + (tid - DSTATE)];
    __syncthreads();
    float dtv = dt[row * DINNER + d];
    float xv  = xc[row * DINNER + d];
    float du = dtv * xv;
    float acc = 0.f;
#pragma unroll
    for (int s = 0; s < DSTATE; ++s) {
      h[s] = expf(dtv * A[s]) * h[s] + du * sB[s];
      acc += h[s] * sC[s];
    }
    y[row * DINNER + d] = acc + Dd * xv;
    __syncthreads();
  }
}

__global__ void gate_silu(const float* __restrict__ y, const float* __restrict__ xz,
                          _Float16* __restrict__ y16) {
  size_t n = (size_t)BB * LTOK * DINNER;
  for (size_t i = (size_t)blockIdx.x * 256 + threadIdx.x; i < n; i += (size_t)gridDim.x * 256) {
    int d = (int)(i % DINNER);
    size_t row = i / DINNER;
    float z = xz[row * (2 * DINNER) + DINNER + d];
    float g = z / (1.f + expf(-z));
    y16[i] = (_Float16)(y[i] * g);
  }
}

// ---------------- host side --------------------------------------------------
static inline void launch_gemm(hipStream_t s, const _Float16* A, int lda,
                               const _Float16* Bt, int ldb, float* C, int ldc,
                               const float* bias, int M, int N, int Kd) {
  int Nt64 = N / 64;
  int tiles = (M / 32) * Nt64;
  gemm_f16<<<(tiles + 7) / 8, 256, 0, s>>>(A, lda, Bt, ldb, C, ldc, bias, Kd, Nt64, tiles);
}

extern "C" void kernel_launch(void* const* d_in, const int* in_sizes, int n_in,
                              void* d_out, int out_size, void* d_ws, size_t ws_size,
                              hipStream_t stream) {
  (void)in_sizes; (void)n_in; (void)out_size; (void)ws_size;
  const float* pts   = (const float*)d_in[0];
  const int*   fps   = (const int*)d_in[1];
  const int*   sidx  = (const int*)d_in[2];
  const float* e_w1  = (const float*)d_in[3];
  const float* e_b1  = (const float*)d_in[4];
  const float* e_g1  = (const float*)d_in[5];
  const float* e_bb1 = (const float*)d_in[6];
  const float* e_m1  = (const float*)d_in[7];
  const float* e_v1  = (const float*)d_in[8];
  const float* e_w2  = (const float*)d_in[9];
  const float* e_b2  = (const float*)d_in[10];
  const float* e_w3  = (const float*)d_in[11];
  const float* e_b3  = (const float*)d_in[12];
  const float* e_g2  = (const float*)d_in[13];
  const float* e_bb2 = (const float*)d_in[14];
  const float* e_m2  = (const float*)d_in[15];
  const float* e_v2  = (const float*)d_in[16];
  const float* e_w4  = (const float*)d_in[17];
  const float* e_b4  = (const float*)d_in[18];
  const float* p_w1  = (const float*)d_in[19];
  const float* p_b1  = (const float*)d_in[20];
  const float* p_w2  = (const float*)d_in[21];
  const float* p_b2  = (const float*)d_in[22];
  const float* normW = (const float*)d_in[23];
  const float* normB = (const float*)d_in[24];
  const float* inW   = (const float*)d_in[25];
  const float* convW = (const float*)d_in[26];
  const float* convB = (const float*)d_in[27];
  const float* xpW   = (const float*)d_in[28];
  const float* dtW   = (const float*)d_in[29];
  const float* dtB   = (const float*)d_in[30];
  const float* Alog  = (const float*)d_in[31];
  const float* Dp    = (const float*)d_in[32];
  const float* outW  = (const float*)d_in[33];
  const float* fW    = (const float*)d_in[34];
  const float* fB    = (const float*)d_in[35];

  char* cur = (char*)d_ws;
  auto alloc = [&](size_t bytes) -> void* {
    void* p = cur; cur += (bytes + 255) & ~(size_t)255; return p;
  };
  _Float16* w2h   = (_Float16*)alloc(sizeof(_Float16) * 256 * 128);
  _Float16* w3h   = (_Float16*)alloc(sizeof(_Float16) * 512 * 512);
  _Float16* w4h   = (_Float16*)alloc(sizeof(_Float16) * 384 * 512);
  _Float16* inWh  = (_Float16*)alloc(sizeof(_Float16) * (size_t)NLAYER * 1536 * 384);
  _Float16* xpWh  = (_Float16*)alloc(sizeof(_Float16) * (size_t)NLAYER * 64 * 768);
  _Float16* outWh = (_Float16*)alloc(sizeof(_Float16) * (size_t)NLAYER * 384 * 768);
  float* center = (float*)alloc(sizeof(float) * BB * GG * 3);
  int*   knn    = (int*)alloc(sizeof(int) * BB * GG * KK);
  float* tokens = (float*)alloc(sizeof(float) * (size_t)BB * GG * DMODEL);
  float* pos    = (float*)alloc(sizeof(float) * (size_t)BB * GG * DMODEL);
  size_t nh = (size_t)BB * LTOK * DMODEL;
  size_t ni = (size_t)BB * LTOK * DINNER;
  float*    hidden = (float*)alloc(sizeof(float) * nh);
  float*    resid  = (float*)alloc(sizeof(float) * nh);
  _Float16* ln16   = (_Float16*)alloc(sizeof(_Float16) * nh);
  float*    xz     = (float*)alloc(sizeof(float) * ni * 2);
  float*    xc     = (float*)alloc(sizeof(float) * ni);
  _Float16* xc16   = (_Float16*)alloc(sizeof(_Float16) * ni);
  float*    dbl    = (float*)alloc(sizeof(float) * (size_t)BB * LTOK * 64);
  float*    dtb    = (float*)alloc(sizeof(float) * ni);
  float*    yb     = (float*)alloc(sizeof(float) * ni);
  _Float16* y16    = (_Float16*)alloc(sizeof(_Float16) * ni);

  auto gs = [](size_t n) { return dim3((unsigned)((n + 255) / 256)); };

  // f32 -> f16 weight staging (x_proj padded 56 -> 64 with zeros)
  cvt_f16_pad<<<gs((size_t)256 * 128), 256, 0, stream>>>(e_w2, w2h, 1, 256, 256, 128);
  cvt_f16_pad<<<gs((size_t)512 * 512), 256, 0, stream>>>(e_w3, w3h, 1, 512, 512, 512);
  cvt_f16_pad<<<gs((size_t)384 * 512), 256, 0, stream>>>(e_w4, w4h, 1, 384, 384, 512);
  cvt_f16_pad<<<gs((size_t)NLAYER * 1536 * 384), 256, 0, stream>>>(inW, inWh, NLAYER, 1536, 1536, 384);
  cvt_f16_pad<<<gs((size_t)NLAYER * 64 * 768), 256, 0, stream>>>(xpW, xpWh, NLAYER, 56, 64, 768);
  cvt_f16_pad<<<gs((size_t)NLAYER * 384 * 768), 256, 0, stream>>>(outW, outWh, NLAYER, 384, 384, 768);

  center_gather<<<gs((size_t)BB * GG), 256, 0, stream>>>(pts, fps, center);
  knn_kernel<<<BB * GG, 256, 0, stream>>>(pts, center, knn);
  encoder_kernel<<<BB * GG, 256, 0, stream>>>(pts, knn, center,
      e_w1, e_b1, e_g1, e_bb1, e_m1, e_v1, w2h, e_b2,
      w3h, e_b3, e_g2, e_bb2, e_m2, e_v2, w4h, e_b4, tokens);
  pos_mlp_kernel<<<BB * GG, 128, 0, stream>>>(center, p_w1, p_b1, p_w2, p_b2, pos);
  reorder_init<<<gs(nh), 256, 0, stream>>>(tokens, pos, sidx, hidden, resid);

  const int M = BB * LTOK;  // 12288
  for (int L = 0; L < NLAYER; ++L) {
    ln_residual<<<BB * LTOK, 128, 0, stream>>>(hidden, resid,
        normW + (size_t)L * DMODEL, normB + (size_t)L * DMODEL, ln16);
    launch_gemm(stream, ln16, DMODEL, inWh + (size_t)L * 1536 * 384, DMODEL,
                xz, 2 * DINNER, nullptr, M, 2 * DINNER, DMODEL);
    conv_silu<<<gs(ni), 256, 0, stream>>>(xz, convW + (size_t)L * DINNER * DCONV,
                                          convB + (size_t)L * DINNER, xc, xc16);
    launch_gemm(stream, xc16, DINNER, xpWh + (size_t)L * 64 * 768, DINNER,
                dbl, 64, nullptr, M, 64, DINNER);
    dt_softplus<<<gs(ni), 256, 0, stream>>>(dbl, dtW + (size_t)L * DINNER * DTRANK,
                                            dtB + (size_t)L * DINNER, dtb);
    scan_kernel<<<BB * 3, 256, 0, stream>>>(dbl, dtb, xc,
        Alog + (size_t)L * DINNER * DSTATE, Dp + (size_t)L * DINNER, yb);
    gate_silu<<<gs(ni), 256, 0, stream>>>(yb, xz, y16);
    launch_gemm(stream, y16, DINNER, outWh + (size_t)L * 384 * 768, DINNER,
                hidden, DMODEL, nullptr, M, DMODEL, DINNER);
  }
  final_ln<<<BB * LTOK, 128, 0, stream>>>(hidden, resid, fW, fB, (float*)d_out);
}
